// AdvancedIFTransformerClassifier_23922967839100
// MI455X (gfx1250) — matrix operand order repaired and checked by reading, not compiled, not run on hardware
//
#include <hip/hip_runtime.h>
#include <hip/hip_bf16.h>
#include <math.h>

// ---------------------------------------------------------------------------
// Types for CDNA5 WMMA (wave32)
// ---------------------------------------------------------------------------
typedef __attribute__((ext_vector_type(16))) __bf16 v16bf;
typedef __attribute__((ext_vector_type(8)))  __bf16 v8bf;
typedef __attribute__((ext_vector_type(8)))  float  v8f;

// Problem constants
#define BB   512
#define TT   96
#define DIN  64
#define HH   256
#define PP   2
#define LL   4
#define HEADS 8
#define EE   768          // 3*H
#define NC   100
#define TP   48           // T/P
#define FF_  25           // rfft bins of 48
#define F2   13           // rfft bins of 25
#define DH   96           // E/HEADS

__device__ __forceinline__ float gelu_exact(float x) {
    return 0.5f * x * (1.0f + erff(x * 0.70710678118654752f));
}

// ---------------------------------------------------------------------------
// Weight pre-swizzle: B f32 [K,N] (ldb) -> bf16 [K/32][N128][32], zero padded.
// Bsw[(kt*N128 + n)*32 + kk] = bf16( B[(kt*32+kk)*ldb + n] )
// Puts each (n, K-tile) column contiguous = the contiguous-k (k = 16h + s) B fragment the GEMM reads.
// ---------------------------------------------------------------------------
__global__ void bswizzle_kernel(const float* __restrict__ B, __bf16* __restrict__ out,
                                int K, int N, int N128, int ldb)
{
    size_t total = (size_t)K * N128;
    size_t idx = (size_t)blockIdx.x * 256 + threadIdx.x;
    if (idx >= total) return;
    int kk = (int)(idx & 31);
    int n  = (int)((idx >> 5) % N128);
    int kt = (int)(idx / ((size_t)N128 * 32));
    float v = (n < N) ? B[(size_t)(kt * 32 + kk) * ldb + n] : 0.0f;
    out[idx] = (__bf16)v;
}

// ---------------------------------------------------------------------------
// bf16-WMMA GEMM:  C[M,N] (+)= A[M,K] * Bsw (+ bias) (+ gelu)
//
// A: f32 activations [M,K] (M % 64 == 0, K % 32 == 0).
// B: pre-swizzled bf16 (see above), N padded to N128 (N128 % 128 == 0).
//
// Block = 256 threads (8 waves). Workgroup tile 64(M) x 128(N), K-step 32.
// Wave w: wr=w>>2 (0..1), wc=w&3 (0..3): 32x32 sub-tile -> 4 WMMA accums.
//
// LDS layouts give every lane a contiguous 32-byte WMMA fragment:
//   As[64][32]: natural K order (no permutation), so lane-group
//               lo reads As[row][lo*16 .. +15]      (k = 16*lo + s, same as Bs)
//   Bs[128][32]: n-major, natural K order; lane reads Bs[col][lo*16 .. +15]
// ---------------------------------------------------------------------------
template<bool ACCUM, bool GELU>
__global__ __launch_bounds__(256)
void gemm_bf16_wmma(const float* __restrict__ A, const __bf16* __restrict__ Bsw,
                    float* __restrict__ C, const float* __restrict__ bias,
                    int M, int N, int N128, int K, int lda, int ldc)
{
    __shared__ __align__(32) __bf16 As[64][32];
    __shared__ __align__(32) __bf16 Bs[128][32];

    const int tid  = threadIdx.x;
    const int lane = tid & 31;
    const int wave = tid >> 5;
    const int wr   = wave >> 2;     // 0..1  (M sub-tile of 32)
    const int wc   = wave & 3;      // 0..3  (N sub-tile of 32)
    const int bm   = blockIdx.y * 64;
    const int bn   = blockIdx.x * 128;

    // A staging: thread -> (row, 8-wide K group), same K dest block (natural order)
    const int ar   = tid >> 2;           // 0..63
    const int acg  = (tid & 3) << 3;     // 0,8,16,24
    const int aq   = acg;
    // B staging: thread -> (n row, 16-elem half)
    const int bnr  = tid & 127;          // 0..127
    const int bsg  = (tid >> 7) << 4;    // 0 or 16

    v8f acc[2][2];
    acc[0][0] = v8f{}; acc[0][1] = v8f{};
    acc[1][0] = v8f{}; acc[1][1] = v8f{};

    const int lo16 = (lane >> 4) * 16;
    const float* aptr = A + (size_t)(bm + ar) * lda + acg;
    const __bf16* bptr = Bsw + ((size_t)bn + bnr) * 32 + bsg;
    const size_t bstep = (size_t)N128 * 32;

    for (int k0 = 0; k0 < K; k0 += 32) {
        // ---- stage A (64x32 f32 -> bf16, natural K order) ----
        {
            float4 a0 = *(const float4*)(aptr + k0);
            float4 a1 = *(const float4*)(aptr + k0 + 4);
            v8bf pk;
            pk[0] = (__bf16)a0.x; pk[1] = (__bf16)a0.y;
            pk[2] = (__bf16)a0.z; pk[3] = (__bf16)a0.w;
            pk[4] = (__bf16)a1.x; pk[5] = (__bf16)a1.y;
            pk[6] = (__bf16)a1.z; pk[7] = (__bf16)a1.w;
            *(v8bf*)&As[ar][aq] = pk;              // one ds_store_b128
        }
        // ---- stage B (128x32 bf16, straight 32-byte copy) ----
        *(v16bf*)&Bs[bnr][bsg] = *(const v16bf*)(bptr + (size_t)(k0 >> 5) * bstep);
        // prefetch next K tile
        if (k0 + 32 < K) {
            __builtin_prefetch(aptr + k0 + 32, 0, 1);
            __builtin_prefetch(bptr + (size_t)((k0 >> 5) + 1) * bstep, 0, 1);
        }
        __syncthreads();

        // ---- fragments + 4 WMMA ----
        v16bf afrag[2], bfrag[2];
        #pragma unroll
        for (int rr = 0; rr < 2; ++rr) {
            int arow = wr * 32 + rr * 16 + (lane & 15);
            afrag[rr] = *(const v16bf*)&As[arow][lo16];
        }
        #pragma unroll
        for (int cc = 0; cc < 2; ++cc) {
            int bcol = wc * 32 + cc * 16 + (lane & 15);
            bfrag[cc] = *(const v16bf*)&Bs[bcol][lo16];
        }
        #pragma unroll
        for (int rr = 0; rr < 2; ++rr)
            #pragma unroll
            for (int cc = 0; cc < 2; ++cc)
                acc[rr][cc] = __builtin_amdgcn_wmma_f32_16x16x32_bf16(
                    false, afrag[rr], false, bfrag[cc], (short)0, acc[rr][cc],
                    false, false);
        __syncthreads();
    }

    // ---- epilogue + store (C/D layout: VGPR r -> M = r + (lane>>4)*8) ----
    #pragma unroll
    for (int rr = 0; rr < 2; ++rr) {
        const int mb = bm + wr * 32 + rr * 16 + (lane >> 4) * 8;
        #pragma unroll
        for (int cc = 0; cc < 2; ++cc) {
            const int nn = bn + wc * 32 + cc * 16 + (lane & 15);
            if (nn >= N) continue;
            #pragma unroll
            for (int r = 0; r < 8; ++r) {
                int m = mb + r;
                float x = acc[rr][cc][r];
                if (bias) x += bias[nn];
                if (GELU) x = gelu_exact(x);
                size_t idx = (size_t)m * ldc + nn;
                if (ACCUM) x += C[idx];
                C[idx] = x;
            }
        }
    }
}

// ---------------------------------------------------------------------------
// Elementwise helpers
// ---------------------------------------------------------------------------
__global__ void add_pe_kernel(float* __restrict__ h, const float* __restrict__ pe,
                              int total)   // h[B,T,H] += pe[t,:]
{
    int i = blockIdx.x * 256 + threadIdx.x;
    if (i >= total) return;
    int c = i % HH;
    int t = (i / HH) % TT;
    h[i] += pe[t * HH + c];
}

__global__ void diff_kernel(const float* __restrict__ h, float* __restrict__ sf,
                            int total)     // sf[b,t,:] = h[b,t,:]-h[b,t-1,:], 0 at t=0
{
    int i = blockIdx.x * 256 + threadIdx.x;
    if (i >= total) return;
    int t = (i / HH) % TT;
    sf[i] = (t == 0) ? 0.0f : (h[i] - h[i - HH]);
}

// real part of rfft over T=48:  f[b,k,c] = sum_t h2[b,t,c] * cos(2*pi*k*t/48)
__global__ void dft_real_kernel(const float* __restrict__ h2, float* __restrict__ f)
{
    int bid = blockIdx.x;          // B * 25
    int b = bid / FF_;
    int kf = bid % FF_;
    int c = threadIdx.x;           // 256 = H
    const float w = 6.28318530717958647f / 48.0f;
    float acc = 0.0f;
    const float* src = h2 + ((size_t)b * TP) * HH + c;
    #pragma unroll 4
    for (int t = 0; t < TP; ++t)
        acc += src[t * HH] * __cosf(w * (float)(kf * t));
    f[((size_t)b * FF_ + kf) * HH + c] = acc;
}

// wt[(dk*I + i)*O + o] = w[o*I*k + i*k + dk]
__global__ void wtrans_kernel(const float* __restrict__ w, float* __restrict__ wt,
                              int O, int I, int k)
{
    int idx = blockIdx.x * 256 + threadIdx.x;
    int total = O * I * k;
    if (idx >= total) return;
    int o  = idx / (I * k);
    int r  = idx % (I * k);
    int i  = r / k;
    int dk = r % k;
    wt[((size_t)(dk * I + i)) * O + o] = w[idx];
}

// im2col over sequence axis: y[b,f, dk*C + c] = x[b, f+dk-lo, c] (zero padded)
__global__ void im2col_kernel(const float* __restrict__ x, float* __restrict__ y,
                              int Bn, int Fdim, int C, int k, int lo)
{
    size_t total = (size_t)Bn * Fdim * k * C;
    size_t idx = (size_t)blockIdx.x * 256 + threadIdx.x;
    if (idx >= total) return;
    int c  = (int)(idx % C);
    int dk = (int)((idx / C) % k);
    int f  = (int)((idx / ((size_t)C * k)) % Fdim);
    int b  = (int)( idx / ((size_t)C * k * Fdim));
    int s  = f + dk - lo;
    y[idx] = (s >= 0 && s < Fdim) ? x[((size_t)b * Fdim + s) * C + c] : 0.0f;
}

// ---------------------------------------------------------------------------
// LayerNorm over E=768; optional second input added first (for SSM residual)
// ---------------------------------------------------------------------------
__global__ __launch_bounds__(256)
void layernorm_kernel(const float* __restrict__ x, const float* __restrict__ addin,
                      const float* __restrict__ g, const float* __restrict__ bta,
                      float* __restrict__ y)
{
    const int row = blockIdx.x;
    const int tid = threadIdx.x;
    const float* xr = x + (size_t)row * EE;
    const float* ar = addin ? addin + (size_t)row * EE : nullptr;
    __shared__ float s1[256], s2[256];

    float vals[3];
    float lsum = 0.0f, lsq = 0.0f;
    #pragma unroll
    for (int j = 0; j < 3; ++j) {
        int i = tid + j * 256;
        float v = xr[i] + (ar ? ar[i] : 0.0f);
        vals[j] = v;
        lsum += v; lsq += v * v;
    }
    s1[tid] = lsum; s2[tid] = lsq;
    __syncthreads();
    for (int s = 128; s > 0; s >>= 1) {
        if (tid < s) { s1[tid] += s1[tid + s]; s2[tid] += s2[tid + s]; }
        __syncthreads();
    }
    const float mean = s1[0] * (1.0f / EE);
    const float var  = s2[0] * (1.0f / EE) - mean * mean;
    const float inv  = rsqrtf(var + 1e-5f);
    #pragma unroll
    for (int j = 0; j < 3; ++j) {
        int i = tid + j * 256;
        y[(size_t)row * EE + i] = (vals[j] - mean) * inv * g[i] + bta[i];
    }
}

// ---------------------------------------------------------------------------
// Fourier attention, one block per (batch, head). T=25, F=13, dh=96.
// ---------------------------------------------------------------------------
__global__ __launch_bounds__(128)
void fourier_attn_kernel(const float* __restrict__ qkv, float* __restrict__ out)
{
    const int bh = blockIdx.x;
    const int b  = bh >> 3;
    const int hd = bh & 7;
    const int tid = threadIdx.x;

    __shared__ float qs[FF_][DH];
    __shared__ float ks[FF_][DH];
    __shared__ float vs[FF_][DH];
    __shared__ float red[128];
    __shared__ float attn_s[16];

    const size_t base = (size_t)b * FF_ * (3 * EE) + (size_t)hd * DH;
    for (int i = tid; i < FF_ * DH; i += 128) {
        int t = i / DH, d = i % DH;
        size_t p = base + (size_t)t * (3 * EE) + d;
        qs[t][d] = qkv[p];
        ks[t][d] = qkv[p + EE];
        vs[t][d] = qkv[p + 2 * EE];
    }
    __syncthreads();

    const float w0 = 6.28318530717958647f / 25.0f;
    const int d = tid;
    float prod[F2];
    if (d < DH) {
        #pragma unroll
        for (int f = 0; f < F2; ++f) {
            float qr = 0, qi = 0, kr = 0, ki = 0;
            for (int t = 0; t < FF_; ++t) {
                float sn, cs;
                __sincosf(w0 * (float)(f * t), &sn, &cs);
                float qv = qs[t][d], kv = ks[t][d];
                qr += qv * cs;  qi -= qv * sn;
                kr += kv * cs;  ki -= kv * sn;
            }
            prod[f] = qr * kr + qi * ki;
        }
    } else {
        #pragma unroll
        for (int f = 0; f < F2; ++f) prod[f] = 0.0f;
    }

    for (int f = 0; f < F2; ++f) {
        red[tid] = (tid < DH) ? prod[f] : 0.0f;
        __syncthreads();
        for (int s = 64; s > 0; s >>= 1) {
            if (tid < s) red[tid] += red[tid + s];
            __syncthreads();
        }
        if (tid == 0) attn_s[f] = red[0] * 0.10206207261596575f; // 96^-0.5
        __syncthreads();
    }

    if (tid == 0) {
        float mx = attn_s[0];
        for (int f = 1; f < F2; ++f) mx = fmaxf(mx, attn_s[f]);
        float sum = 0.0f;
        for (int f = 0; f < F2; ++f) { attn_s[f] = __expf(attn_s[f] - mx); sum += attn_s[f]; }
        float r = 1.0f / sum;
        for (int f = 0; f < F2; ++f) attn_s[f] *= r;
    }
    __syncthreads();

    if (d < DH) {
        float vfr[F2], vfi[F2];
        #pragma unroll
        for (int f = 0; f < F2; ++f) {
            float ar = 0, ai = 0;
            for (int t = 0; t < FF_; ++t) {
                float sn, cs;
                __sincosf(w0 * (float)(f * t), &sn, &cs);
                float vv = vs[t][d];
                ar += vv * cs;  ai -= vv * sn;
            }
            vfr[f] = ar * attn_s[f];
            vfi[f] = ai * attn_s[f];
        }
        for (int t = 0; t < FF_; ++t) {
            float acc = vfr[0];
            #pragma unroll
            for (int f = 1; f < F2; ++f) {
                float sn, cs;
                __sincosf(w0 * (float)(f * t), &sn, &cs);
                acc += 2.0f * (vfr[f] * cs - vfi[f] * sn);
            }
            out[((size_t)b * FF_ + t) * EE + hd * DH + d] = acc * (1.0f / 25.0f);
        }
    }
}

// mean over the 25 frequency positions
__global__ void meanpool_kernel(const float* __restrict__ x, float* __restrict__ y)
{
    int b = blockIdx.x;
    for (int c = threadIdx.x; c < EE; c += blockDim.x) {
        float s = 0.0f;
        for (int t = 0; t < FF_; ++t) s += x[((size_t)b * FF_ + t) * EE + c];
        y[(size_t)b * EE + c] = s * (1.0f / 25.0f);
    }
}

// ---------------------------------------------------------------------------
// Host-side orchestration
// ---------------------------------------------------------------------------
static inline dim3 gemm_grid(int M, int N128) { return dim3(N128 / 128, M / 64); }
static inline int  pad128(int n) { return (n + 127) & ~127; }

static inline void swz(const float* B, __bf16* out, int K, int N, int ldb,
                       hipStream_t stream)
{
    size_t total = (size_t)K * pad128(N);
    bswizzle_kernel<<<(int)((total + 255) / 256), 256, 0, stream>>>(
        B, out, K, N, pad128(N), ldb);
}

extern "C" void kernel_launch(void* const* d_in, const int* in_sizes, int n_in,
                              void* d_out, int out_size, void* d_ws, size_t ws_size,
                              hipStream_t stream)
{
    (void)in_sizes; (void)n_in; (void)out_size; (void)ws_size;
    const float* x       = (const float*)d_in[0];
    const float* W_in    = (const float*)d_in[1];
    const float* b_in    = (const float*)d_in[2];
    const float* pe      = (const float*)d_in[3];
    const float* W_shape = (const float*)d_in[4];
    const float* b_shape = (const float*)d_in[5];
    const float* W_patch = (const float*)d_in[6];
    const float* b_patch = (const float*)d_in[7];
    const float* conv_w1 = (const float*)d_in[8];
    const float* conv_b1 = (const float*)d_in[9];
    const float* conv_w2 = (const float*)d_in[10];
    const float* conv_b2 = (const float*)d_in[11];
    const float* conv_w4 = (const float*)d_in[12];
    const float* conv_b4 = (const float*)d_in[13];
    const float* ln1_g   = (const float*)d_in[14];
    const float* ln1_b   = (const float*)d_in[15];
    const float* Wqkv    = (const float*)d_in[16];
    const float* Wo      = (const float*)d_in[17];
    const float* bo      = (const float*)d_in[18];
    const float* ln2_g   = (const float*)d_in[19];
    const float* ln2_b   = (const float*)d_in[20];
    const float* Wf1     = (const float*)d_in[21];
    const float* bf1     = (const float*)d_in[22];
    const float* Wf2     = (const float*)d_in[23];
    const float* bf2     = (const float*)d_in[24];
    const float* ssm_w   = (const float*)d_in[25];
    const float* ssm_b   = (const float*)d_in[26];
    const float* ssm_g   = (const float*)d_in[27];
    const float* ssm_bn  = (const float*)d_in[28];
    const float* W_out   = (const float*)d_in[29];
    const float* b_out   = (const float*)d_in[30];
    float* out = (float*)d_out;

    // ---- workspace layout ----
    float* ws = (float*)d_ws;
    size_t off = 0;
    auto alloc  = [&](size_t n) { float* p = ws + off; off += n; return p; };
    auto allocb = [&](size_t n) { __bf16* p = (__bf16*)(ws + off); off += (n + 1) / 2; return p; };

    float* h1    = alloc((size_t)BB * TT * HH);          // [B,96,256]
    float* sf    = alloc((size_t)BB * TT * HH);
    float* h2    = alloc((size_t)BB * TP * HH);          // [B,48,256]
    float* fre   = alloc((size_t)BB * FF_ * HH);         // [B,25,256]
    float* hcat  = alloc((size_t)BB * FF_ * EE);         // [B,25,768]
    float* ybuf  = alloc((size_t)BB * FF_ * EE);
    float* abuf  = alloc((size_t)BB * FF_ * EE);
    float* big   = alloc((size_t)BB * FF_ * 4 * EE);     // qkv / ff1 / im2col
    float* pooled= alloc((size_t)BB * EE);
    float* wt1   = alloc((size_t)HH * HH);               // conv weights, f32 transposed
    float* wt2   = alloc((size_t)2 * HH * HH);
    float* wt4   = alloc((size_t)4 * HH * HH);
    float* wts   = alloc((size_t)3 * EE * EE);

    // pre-swizzled bf16 weights (N padded to 128)
    __bf16* Win_s  = allocb((size_t)DIN * 256);
    __bf16* Wsh_s  = allocb((size_t)HH * 256);
    __bf16* Wpa_s  = allocb((size_t)PP * HH * 256);
    __bf16* wt1_s  = allocb((size_t)HH * 256);
    __bf16* wt2_s  = allocb((size_t)2 * HH * 256);
    __bf16* wt4_s  = allocb((size_t)4 * HH * 256);
    __bf16* wts_s  = allocb((size_t)3 * EE * 768);
    __bf16* Wqkv_s = allocb((size_t)LL * EE * 2304);
    __bf16* Wo_s   = allocb((size_t)LL * EE * 768);
    __bf16* Wf1_s  = allocb((size_t)LL * EE * 3072);
    __bf16* Wf2_s  = allocb((size_t)LL * 4 * EE * 768);
    __bf16* Wout_s = allocb((size_t)EE * 128);

    const int MT = BB * TT;        // 49152
    const int MP = BB * TP;        // 24576
    const int MF = BB * FF_;       // 12800

    // ---- weight prep: conv transposes + all swizzles ----
    wtrans_kernel<<<(HH * HH * 1 + 255) / 256, 256, 0, stream>>>(conv_w1, wt1, HH, HH, 1);
    wtrans_kernel<<<(HH * HH * 2 + 255) / 256, 256, 0, stream>>>(conv_w2, wt2, HH, HH, 2);
    wtrans_kernel<<<(HH * HH * 4 + 255) / 256, 256, 0, stream>>>(conv_w4, wt4, HH, HH, 4);
    wtrans_kernel<<<(EE * EE * 3 + 255) / 256, 256, 0, stream>>>(ssm_w, wts, EE, EE, 3);
    swz(W_in,    Win_s, DIN,      HH, HH, stream);
    swz(W_shape, Wsh_s, HH,       HH, HH, stream);
    swz(W_patch, Wpa_s, PP * HH,  HH, HH, stream);
    swz(wt1,     wt1_s, HH,       HH, HH, stream);
    swz(wt2,     wt2_s, 2 * HH,   HH, HH, stream);
    swz(wt4,     wt4_s, 4 * HH,   HH, HH, stream);
    swz(wts,     wts_s, 3 * EE,   EE, EE, stream);
    for (int i = 0; i < LL; ++i) {
        swz(Wqkv + (size_t)i * EE * 3 * EE, Wqkv_s + (size_t)i * EE * 2304,
            EE, 3 * EE, 3 * EE, stream);
        swz(Wo   + (size_t)i * EE * EE,     Wo_s   + (size_t)i * EE * 768,
            EE, EE, EE, stream);
        swz(Wf1  + (size_t)i * EE * 4 * EE, Wf1_s  + (size_t)i * EE * 3072,
            EE, 4 * EE, 4 * EE, stream);
        swz(Wf2  + (size_t)i * 4 * EE * EE, Wf2_s  + (size_t)i * 4 * EE * 768,
            4 * EE, EE, EE, stream);
    }
    swz(W_out, Wout_s, EE, NC, NC, stream);

    // 1. input projection: h1 = x @ W_in + b_in
    gemm_bf16_wmma<false,false><<<gemm_grid(MT, 256), 256, 0, stream>>>(
        x, Win_s, h1, b_in, MT, HH, 256, DIN, DIN, HH);
    // 2. positional embedding
    add_pe_kernel<<<(MT * HH + 255) / 256, 256, 0, stream>>>(h1, pe, MT * HH);
    // 3. shape-aware: sf = diff(h1), h1 += sf @ W_shape + b_shape
    diff_kernel<<<(MT * HH + 255) / 256, 256, 0, stream>>>(h1, sf, MT * HH);
    gemm_bf16_wmma<true,false><<<gemm_grid(MT, 256), 256, 0, stream>>>(
        sf, Wsh_s, h1, b_shape, MT, HH, 256, HH, HH, HH);
    // 4. patch embedding: [B,48,512] @ W_patch -> h2 [B,48,256]
    gemm_bf16_wmma<false,false><<<gemm_grid(MP, 256), 256, 0, stream>>>(
        h1, Wpa_s, h2, b_patch, MP, HH, 256, PP * HH, PP * HH, HH);
    // 5. rfft real part over T=48 -> fre [B,25,256]
    dft_real_kernel<<<BB * FF_, HH, 0, stream>>>(h2, fre);
    // 6. multiscale convs -> hcat[:, :, 0:256 | 256:512 | 512:768]
    gemm_bf16_wmma<false,false><<<gemm_grid(MF, 256), 256, 0, stream>>>(
        fre, wt1_s, hcat + 0, conv_b1, MF, HH, 256, HH, HH, EE);
    im2col_kernel<<<(int)(((size_t)MF * 2 * HH + 255) / 256), 256, 0, stream>>>(
        fre, big, BB, FF_, HH, 2, 0);
    gemm_bf16_wmma<false,false><<<gemm_grid(MF, 256), 256, 0, stream>>>(
        big, wt2_s, hcat + HH, conv_b2, MF, HH, 256, 2 * HH, 2 * HH, EE);
    im2col_kernel<<<(int)(((size_t)MF * 4 * HH + 255) / 256), 256, 0, stream>>>(
        fre, big, BB, FF_, HH, 4, 1);
    gemm_bf16_wmma<false,false><<<gemm_grid(MF, 256), 256, 0, stream>>>(
        big, wt4_s, hcat + 2 * HH, conv_b4, MF, HH, 256, 4 * HH, 4 * HH, EE);

    // 7. transformer layers
    for (int i = 0; i < LL; ++i) {
        layernorm_kernel<<<MF, 256, 0, stream>>>(
            hcat, nullptr, ln1_g + i * EE, ln1_b + i * EE, ybuf);
        gemm_bf16_wmma<false,false><<<gemm_grid(MF, 2304), 256, 0, stream>>>(
            ybuf, Wqkv_s + (size_t)i * EE * 2304, big, nullptr,
            MF, 3 * EE, 2304, EE, EE, 3 * EE);
        fourier_attn_kernel<<<BB * HEADS, 128, 0, stream>>>(big, abuf);
        gemm_bf16_wmma<true,false><<<gemm_grid(MF, 768), 256, 0, stream>>>(
            abuf, Wo_s + (size_t)i * EE * 768, hcat, bo + i * EE,
            MF, EE, 768, EE, EE, EE);
        layernorm_kernel<<<MF, 256, 0, stream>>>(
            hcat, nullptr, ln2_g + i * EE, ln2_b + i * EE, ybuf);
        gemm_bf16_wmma<false,true><<<gemm_grid(MF, 3072), 256, 0, stream>>>(
            ybuf, Wf1_s + (size_t)i * EE * 3072, big, bf1 + i * 4 * EE,
            MF, 4 * EE, 3072, EE, EE, 4 * EE);
        gemm_bf16_wmma<true,false><<<gemm_grid(MF, 768), 256, 0, stream>>>(
            big, Wf2_s + (size_t)i * 4 * EE * 768, hcat, bf2 + i * EE,
            MF, EE, 768, 4 * EE, 4 * EE, EE);
    }

    // 8. SSM conv (k=3, same pad) as im2col + GEMM -> ybuf
    im2col_kernel<<<(int)(((size_t)MF * 3 * EE + 255) / 256), 256, 0, stream>>>(
        hcat, big, BB, FF_, EE, 3, 1);
    gemm_bf16_wmma<false,false><<<gemm_grid(MF, 768), 256, 0, stream>>>(
        big, wts_s, ybuf, ssm_b, MF, EE, 768, 3 * EE, 3 * EE, EE);
    // 9. final LN of (hcat + ybuf) -> abuf
    layernorm_kernel<<<MF, 256, 0, stream>>>(hcat, ybuf, ssm_g, ssm_bn, abuf);
    // 10. mean over positions, classifier head
    meanpool_kernel<<<BB, 256, 0, stream>>>(abuf, pooled);
    gemm_bf16_wmma<false,false><<<gemm_grid(BB, 128), 256, 0, stream>>>(
        pooled, Wout_s, out, b_out, BB, NC, 128, EE, EE, NC);
}
